// MultitaskGatedCTM_33079838114486
// MI455X (gfx1250) — matrix operand order repaired and checked by reading, not compiled, not run on hardware
//
#include <hip/hip_runtime.h>

typedef __attribute__((ext_vector_type(16))) _Float16 v16h;
typedef __attribute__((ext_vector_type(8)))  _Float16 v8h;
typedef __attribute__((ext_vector_type(8)))  float    v8f;

#define BB    128
#define NN    2048
#define MM    32
#define TRS   48      // linear trace slots: 32 initial + 16 ticks
#define TT    16
#define RR    512
#define SOO   512
#define SETS4 4

__device__ __forceinline__ float sigm(float x) { return 1.0f / (1.0f + expf(-x)); }

// A-fragment (16x32 f16, ISA 7.12.2 layout) from f32 row-major source.
// p must point at A + row*lda + lh*8 (16B aligned); runs are [k0, k0+8) and [k0+16, k0+24).
__device__ __forceinline__ v16h load_afrag_f32(const float* __restrict__ p) {
  float4 q0 = *(const float4*)(p);
  float4 q1 = *(const float4*)(p + 4);
  float4 q2 = *(const float4*)(p + 16);
  float4 q3 = *(const float4*)(p + 20);
  v16h a;
  a[0]  = (_Float16)q0.x; a[1]  = (_Float16)q0.y; a[2]  = (_Float16)q0.z; a[3]  = (_Float16)q0.w;
  a[4]  = (_Float16)q1.x; a[5]  = (_Float16)q1.y; a[6]  = (_Float16)q1.z; a[7]  = (_Float16)q1.w;
  a[8]  = (_Float16)q2.x; a[9]  = (_Float16)q2.y; a[10] = (_Float16)q2.z; a[11] = (_Float16)q2.w;
  a[12] = (_Float16)q3.x; a[13] = (_Float16)q3.y; a[14] = (_Float16)q3.z; a[15] = (_Float16)q3.w;
  return a;
}

// B-fragment: f16 K-runs [p, p+8) and [p+16, p+24) of one W row (16B aligned).
__device__ __forceinline__ v16h load_bfrag_f16(const _Float16* __restrict__ p) {
  v8h lo = *(const v8h*)(p);
  v8h hi = *(const v8h*)(p + 16);
  v16h b;
#pragma unroll
  for (int i = 0; i < 8; ++i) { b[i] = lo[i]; b[8 + i] = hi[i]; }
  return b;
}

// ---------------------------------------------------------------------------
// WMMA GEMM: C[M,N] = act(A[M,K] @ W[N,K]^T + bias). f32 A, f16 W, f32 out.
// Requires M%32==0, N%64==0, K%32==0. One wave per 32-row x 64-col tile group:
// per K step, 2 A fragments + 4 B fragments feed 8 consecutive v_wmma
// (B reused across M, A reused across N).
// ---------------------------------------------------------------------------
__global__ void gemm_wmma_kernel(const float* __restrict__ A, int lda,
                                 const _Float16* __restrict__ W, int ldw,
                                 const float* __restrict__ bias,
                                 float* __restrict__ C, int ldc,
                                 int M, int N, int K, int actmode) {
  int ng = N >> 6;
  int tile = blockIdx.x * blockDim.y + threadIdx.y;
  if (tile >= (M >> 5) * ng) return;
  int m0 = (tile / ng) << 5;
  int n0 = (tile % ng) << 6;
  int l  = threadIdx.x;
  int lh = l >> 4;
  int ll = l & 15;

  const float* Ap0 = A + (size_t)(m0 + ll) * lda + lh * 8;
  const float* Ap1 = Ap0 + (size_t)16 * lda;
  const _Float16* Wp = W + (size_t)(n0 + ll) * ldw + lh * 8;
  const size_t wstride = (size_t)16 * ldw;   // advance one 16-col tile

  v8f acc[2][4] = {{{}, {}, {}, {}}, {{}, {}, {}, {}}};
  for (int k0 = 0; k0 < K; k0 += 32) {
    __builtin_prefetch(Wp + k0 + 256, 0, 1);
    v16h av[2];
    av[0] = load_afrag_f32(Ap0 + k0);
    av[1] = load_afrag_f32(Ap1 + k0);
    v16h bv[4];
#pragma unroll
    for (int j = 0; j < 4; ++j) bv[j] = load_bfrag_f16(Wp + j * wstride + k0);
#pragma unroll
    for (int mi = 0; mi < 2; ++mi)
#pragma unroll
      for (int j = 0; j < 4; ++j)
        acc[mi][j] = __builtin_amdgcn_wmma_f32_16x16x32_f16(false, av[mi], false, bv[j],
                                                            (short)0, acc[mi][j], false, false);
  }
#pragma unroll
  for (int mi = 0; mi < 2; ++mi) {
#pragma unroll
    for (int j = 0; j < 4; ++j) {
      int n = n0 + j * 16 + ll;
      float bsv = bias[n];
#pragma unroll
      for (int v = 0; v < 8; ++v) {
        float val = acc[mi][j][v] + bsv;
        if (actmode == 1) val = fmaxf(val, 0.0f);
        C[(size_t)(m0 + mi * 16 + v + (lh << 3)) * ldc + n] = val;
      }
    }
  }
}

static void gemm(hipStream_t s, const float* A, int lda, const _Float16* W, int ldw,
                 const float* bias, float* C, int ldc, int M, int N, int K, int actmode) {
  int tiles = (M >> 5) * (N >> 6);
  dim3 blk(32, 4);
  dim3 grd((tiles + 3) / 4);
  gemm_wmma_kernel<<<grd, blk, 0, s>>>(A, lda, W, ldw, bias, C, ldc, M, N, K, actmode);
}

// ---------------------------------------------------------------------------
// Per-neuron model. trace16: (B, N, 48) f16 linear slots; read window starts at rb.
// One wave per (neuron, 16-batch tile): 1 A fragment -> 16 WMMAs -> GLU + dot.
// ---------------------------------------------------------------------------
__global__ void neuron_kernel(const _Float16* __restrict__ trace16,
                              const _Float16* __restrict__ w1h,
                              const float* __restrict__ b1,
                              const float* __restrict__ w2,
                              const float* __restrict__ b2,
                              float* __restrict__ actbuf, int rb) {
  __shared__ float h2s[16][257];
  int n  = blockIdx.x >> 3;
  int b0 = (blockIdx.x & 7) << 4;
  int l  = threadIdx.x;
  int lh = l >> 4;
  int ll = l & 15;

  const _Float16* tp = trace16 + ((size_t)(b0 + ll) * NN + n) * TRS + rb + lh * 8;
  v16h a;
#pragma unroll
  for (int i = 0; i < 8; ++i) { a[i] = tp[i]; a[8 + i] = tp[16 + i]; }

  const _Float16* w1n = w1h + ((size_t)n * 256 + ll) * MM + lh * 8;
  const float* b1n = b1 + (size_t)n * 256;

  for (int ct = 0; ct < 16; ++ct) {
    int c0 = ct << 4;
    v16h b = load_bfrag_f16(w1n + (size_t)c0 * MM);
    v8f acc = {};
    acc = __builtin_amdgcn_wmma_f32_16x16x32_f16(false, a, false, b,
                                                 (short)0, acc, false, false);
#pragma unroll
    for (int v = 0; v < 8; ++v)
      h2s[v + (lh << 3)][c0 + ll] = acc[v] + b1n[c0 + ll];
  }
  __syncthreads();
  if (l < 16) {
    const float* w2n = w2 + (size_t)n * 128;
    float s = 0.0f;
    for (int c = 0; c < 128; ++c) {
      float gv = h2s[l][c] * sigm(h2s[l][c + 128]);
      s += gv * w2n[c];
    }
    actbuf[(size_t)(b0 + l) * NN + n] = s + b2[n];
  }
}

// ---------------------------------------------------------------------------
// Encoder helpers
// ---------------------------------------------------------------------------
__global__ void conv3x3_bn_relu(const float* __restrict__ in,
                                const float* __restrict__ w,
                                const float* __restrict__ cb,
                                const float* __restrict__ g,
                                const float* __restrict__ bb,
                                float* __restrict__ out,
                                int Cin, int Cout, int H, int W, int total) {
  int idx = blockIdx.x * blockDim.x + threadIdx.x;
  if (idx >= total) return;
  int x = idx % W; int t = idx / W;
  int y = t % H;   t /= H;
  int co = t % Cout; int b = t / Cout;
  float acc = cb[co];
  for (int ci = 0; ci < Cin; ++ci) {
    const float* ip = in + ((size_t)(b * Cin + ci) * H) * W;
    const float* wp = w + ((size_t)(co * Cin + ci) * 9);
    for (int ky = 0; ky < 3; ++ky) {
      int iy = y + ky - 1;
      if (iy < 0 || iy >= H) continue;
      for (int kx = 0; kx < 3; ++kx) {
        int ix = x + kx - 1;
        if (ix < 0 || ix >= W) continue;
        acc += ip[(size_t)iy * W + ix] * wp[ky * 3 + kx];
      }
    }
  }
  float scale = g[co] * rsqrtf(1.0f + 1e-5f);
  out[idx] = fmaxf(acc * scale + bb[co], 0.0f);
}

__global__ void maxpool2_kernel(const float* __restrict__ in, float* __restrict__ out,
                                int C, int H, int W, int total) {
  int idx = blockIdx.x * blockDim.x + threadIdx.x;
  if (idx >= total) return;
  int H2 = H >> 1, W2 = W >> 1;
  int x = idx % W2; int t = idx / W2;
  int y = t % H2;   t /= H2;
  int c = t % C;    int b = t / C;
  const float* ip = in + ((size_t)(b * C + c) * H) * W;
  float v = ip[(size_t)(2 * y) * W + 2 * x];
  v = fmaxf(v, ip[(size_t)(2 * y) * W + 2 * x + 1]);
  v = fmaxf(v, ip[(size_t)(2 * y + 1) * W + 2 * x]);
  v = fmaxf(v, ip[(size_t)(2 * y + 1) * W + 2 * x + 1]);
  out[idx] = v;
}

__global__ void avgpool_flat_kernel(const float* __restrict__ in, float* __restrict__ out,
                                    int total) {
  int idx = blockIdx.x * blockDim.x + threadIdx.x;
  if (idx >= total) return;
  int o = idx % 256; int b = idx / 256;
  int c = o >> 2; int p = (o >> 1) & 1; int q = o & 1;
  const float* ip = in + ((size_t)(b * 64 + c) * 8) * 8;
  float acc = 0.0f;
  for (int dy = 0; dy < 4; ++dy)
    for (int dx = 0; dx < 4; ++dx)
      acc += ip[(size_t)(p * 4 + dy) * 8 + q * 4 + dx];
  out[idx] = acc * (1.0f / 16.0f);
}

// ---------------------------------------------------------------------------
// Elementwise / small kernels
// ---------------------------------------------------------------------------
__global__ void f2h_kernel(const float* __restrict__ s, _Float16* __restrict__ d, int total) {
  int idx = blockIdx.x * blockDim.x + threadIdx.x;
  if (idx >= total) return;
  d[idx] = (_Float16)s[idx];
}

__global__ void glu_kernel(const float* __restrict__ in, float* __restrict__ out,
                           int D, int total) {
  int idx = blockIdx.x * blockDim.x + threadIdx.x;
  if (idx >= total) return;
  int r = idx / D, j = idx % D;
  float av = in[(size_t)r * 2 * D + j];
  float bv = in[(size_t)r * 2 * D + D + j];
  out[idx] = av * sigm(bv);
}

__global__ void layernorm_kernel(const float* __restrict__ in, float* __restrict__ out,
                                 const float* __restrict__ g, const float* __restrict__ bta,
                                 int D, int relu) {
  int row = blockIdx.x;
  const float* x = in + (size_t)row * D;
  float* y = out + (size_t)row * D;
  __shared__ float r1[256];
  __shared__ float r2[256];
  float s = 0.0f, s2 = 0.0f;
  for (int i = threadIdx.x; i < D; i += blockDim.x) { float v = x[i]; s += v; s2 += v * v; }
  r1[threadIdx.x] = s; r2[threadIdx.x] = s2;
  __syncthreads();
  for (int off = 128; off > 0; off >>= 1) {
    if (threadIdx.x < off) { r1[threadIdx.x] += r1[threadIdx.x + off]; r2[threadIdx.x] += r2[threadIdx.x + off]; }
    __syncthreads();
  }
  float mu  = r1[0] / D;
  float var = r2[0] / D - mu * mu;
  float rs  = rsqrtf(var + 1e-5f);
  for (int i = threadIdx.x; i < D; i += blockDim.x) {
    float v = (x[i] - mu) * rs * g[i] + bta[i];
    if (relu) v = fmaxf(v, 0.0f);
    y[i] = v;
  }
}

__global__ void bcast_rows_kernel(const float* __restrict__ src, float* __restrict__ dst,
                                  int cols, int total) {
  int idx = blockIdx.x * blockDim.x + threadIdx.x;
  if (idx >= total) return;
  dst[idx] = src[idx % cols];
}

__global__ void copy_to_pre_kernel(const float* __restrict__ src, float* __restrict__ pre,
                                   int cols, int dstoff, int total) {
  int idx = blockIdx.x * blockDim.x + threadIdx.x;
  if (idx >= total) return;
  int b = idx / cols, j = idx % cols;
  pre[(size_t)b * (RR + NN) + dstoff + j] = src[idx];
}

// init linear trace slots 0..31 with start_trace (broadcast over batch), f16
__global__ void trace_init_kernel(const float* __restrict__ st, _Float16* __restrict__ tr,
                                  int total) {
  int idx = blockIdx.x * blockDim.x + threadIdx.x;
  if (idx >= total) return;
  int m = idx & 31;
  size_t bn = (size_t)idx >> 5;
  int n = (int)(bn & (NN - 1));
  tr[bn * TRS + m] = (_Float16)st[(n << 5) + m];
}

__global__ void trace_store_kernel(const float* __restrict__ preact,
                                   _Float16* __restrict__ tr, int slot, int total) {
  int idx = blockIdx.x * blockDim.x + threadIdx.x;
  if (idx >= total) return;
  tr[(size_t)idx * TRS + slot] = (_Float16)preact[idx];
}

__global__ void gate_softmax_kernel(const float* __restrict__ gh,
                                    const float* __restrict__ w2,
                                    const float* __restrict__ b2,
                                    float* __restrict__ gw) {
  int b = blockIdx.x * blockDim.x + threadIdx.x;
  if (b >= BB) return;
  float sv[SETS4];
  float mx = -1e30f;
  for (int s = 0; s < SETS4; ++s) {
    float acc = b2[s];
    for (int k = 0; k < 64; ++k) acc += gh[(size_t)b * 64 + k] * w2[s * 64 + k];
    sv[s] = acc; mx = fmaxf(mx, acc);
  }
  float sum = 0.0f;
  for (int s = 0; s < SETS4; ++s) { sv[s] = expf(sv[s] - mx); sum += sv[s]; }
  for (int s = 0; s < SETS4; ++s) gw[(size_t)b * SETS4 + s] = sv[s] / sum;
}

__global__ void sync_out_kernel(const float* __restrict__ actbuf,
                                const int* __restrict__ il, const int* __restrict__ ir,
                                const float* __restrict__ decay,
                                const float* __restrict__ gw,
                                float* __restrict__ aS, float* __restrict__ bS,
                                float* __restrict__ syncO, int S, int total) {
  int idx = blockIdx.x * blockDim.x + threadIdx.x;
  if (idx >= total) return;
  int b = idx / S, j = idx % S;
  float acc = 0.0f;
  for (int s = 0; s < SETS4; ++s) {
    int o = s * S + j;
    float r  = expf(-fminf(fmaxf(decay[o], 0.0f), 15.0f));
    float pp = actbuf[(size_t)b * NN + il[o]] * actbuf[(size_t)b * NN + ir[o]];
    size_t st = ((size_t)b * SETS4 + s) * S + j;
    float av = r * aS[st] + pp;
    float bv = r * bS[st] + 1.0f;
    aS[st] = av; bS[st] = bv;
    acc += av * rsqrtf(bv + 1e-8f) * gw[(size_t)b * SETS4 + s];
  }
  syncO[(size_t)b * S + j] = acc;
}

// final ragged task projection (N = 10 / 100, K = 64): plain VALU
__global__ void head_out_kernel(const float* __restrict__ in,
                                const float* __restrict__ w3,
                                const float* __restrict__ b3,
                                float* __restrict__ out, int osz, int total) {
  int idx = blockIdx.x * blockDim.x + threadIdx.x;
  if (idx >= total) return;
  int r = idx / osz, o = idx % osz;
  float acc = b3[o];
  for (int k = 0; k < 64; ++k) acc += in[(size_t)r * 64 + k] * w3[o * 64 + k];
  out[idx] = acc;
}

__global__ void balance_kernel(const float* __restrict__ gwA,
                               const float* __restrict__ gwO,
                               float* __restrict__ out) {
  __shared__ float acc[TT];
  int t = threadIdx.x;
  if (t < TT) {
    const float* gws[2] = {gwA, gwO};
    float cv[2];
    for (int q = 0; q < 2; ++q) {
      float mu[SETS4] = {0, 0, 0, 0};
      for (int b = 0; b < BB; ++b)
        for (int s = 0; s < SETS4; ++s)
          mu[s] += gws[q][((size_t)t * BB + b) * SETS4 + s];
      float m = 0.0f;
      for (int s = 0; s < SETS4; ++s) { mu[s] /= (float)BB; m += mu[s]; }
      m /= (float)SETS4;
      float var = 0.0f;
      for (int s = 0; s < SETS4; ++s) { float d = mu[s] - m; var += d * d; }
      var /= (float)(SETS4 - 1);
      cv[q] = sqrtf(var) / (m + 1e-8f);
    }
    acc[t] = 0.5f * (cv[0] + cv[1]);
  }
  __syncthreads();
  if (t == 0) {
    float s = 0.0f;
    for (int i = 0; i < TT; ++i) s += acc[i];
    out[0] = s / (float)TT;
  }
}

// ---------------------------------------------------------------------------
// Host orchestration
// ---------------------------------------------------------------------------
#define LAUNCH1D(kern, total, ...) kern<<<((total) + 255) / 256, 256, 0, stream>>>(__VA_ARGS__)

extern "C" void kernel_launch(void* const* d_in, const int* in_sizes, int n_in,
                              void* d_out, int out_size, void* d_ws, size_t ws_size,
                              hipStream_t stream) {
  (void)in_sizes; (void)n_in; (void)out_size; (void)ws_size;
  auto F = [&](int i) { return (const float*)d_in[i]; };
  auto I = [&](int i) { return (const int*)d_in[i]; };

  // input index map: 0:x, conv/bn 1..16, efc 17..20, kv 21..24, q 25..26 (dead),
  // mha 27..30, syn 31..34, nm 35..38, decay 39..40, gates 41..48, start 49..50,
  // tasks 51..70, idx 71..74
  const float* x = F(0);
  const float* syn_b = F(32);
  const float* syn_ln_g = F(33); const float* syn_ln_b = F(34);
  const float* nm_b1 = F(36); const float* nm_w2 = F(37); const float* nm_b2 = F(38);
  const float* decay_o = F(40);
  const float* gA1_b = F(42); const float* gA2_w = F(43); const float* gA2_b = F(44);
  const float* gO1_b = F(46); const float* gO2_w = F(47); const float* gO2_b = F(48);
  const float* start_act = F(49); const float* start_trace = F(50);
  const int* idx_out_l = I(73); const int* idx_out_r = I(74);

  // ---- workspace: f32 region then f16 region (all sizes multiples of 16 elems) ----
  float* base = (float*)d_ws;
  size_t off = 0;
  auto WS = [&](size_t n) { float* p = base + off; off += n; return p; };
  float* bufA   = WS((size_t)BB * 16 * 32 * 32);   // conv scratch / hglu / ht2
  float* bufB   = WS((size_t)BB * 32 * 32 * 32);   // conv scratch / synout
  float* bufC   = WS((size_t)BB * 32 * 16 * 16);   // conv scratch / pre
  float* bufD   = WS((size_t)BB * 64 * 8 * 8);     // conv scratch / preact
  float* bufE   = WS((size_t)BB * 64 * 8 * 8);     // conv scratch / ht1
  float* flat   = WS((size_t)BB * 256);
  float* tmp256 = WS((size_t)BB * 256);
  float* gluo   = WS((size_t)BB * 128);
  float* feat   = WS((size_t)BB * RR);
  float* kvt    = WS((size_t)BB * RR);
  float* kvb    = WS((size_t)BB * RR);
  float* vh     = WS((size_t)BB * RR);
  float* attn   = WS((size_t)BB * RR);
  float* actb   = WS((size_t)BB * NN);
  float* aO     = WS((size_t)BB * SETS4 * SOO);
  float* bO     = WS((size_t)BB * SETS4 * SOO);
  float* gh     = WS((size_t)BB * 64);
  float* gwA_t  = WS((size_t)TT * BB * SETS4);
  float* gwO_t  = WS((size_t)TT * BB * SETS4);
  float* syncO_t= WS((size_t)TT * BB * SOO);

  _Float16* hbase = (_Float16*)(base + off);
  size_t hoff = 0;
  auto HS = [&](size_t n) { _Float16* p = hbase + hoff; hoff += n; return p; };
  _Float16* efc1h  = HS((size_t)256 * 256);
  _Float16* efc2h  = HS((size_t)RR * 128);
  _Float16* kvh    = HS((size_t)RR * RR);
  _Float16* wvh    = HS((size_t)RR * RR);
  _Float16* mouth  = HS((size_t)RR * RR);
  _Float16* synh   = HS((size_t)2 * NN * (RR + NN));
  _Float16* gA1h   = HS((size_t)64 * NN);
  _Float16* gO1h   = HS((size_t)64 * NN);
  _Float16* nmw1h  = HS((size_t)NN * 256 * MM);
  _Float16* tw1h[2]; _Float16* tw2h[2];
  tw1h[0] = HS((size_t)256 * SOO); tw2h[0] = HS((size_t)64 * 256);
  tw1h[1] = HS((size_t)256 * SOO); tw2h[1] = HS((size_t)64 * 256);
  _Float16* trace16 = HS((size_t)BB * NN * TRS);

  // aliases (lifetimes disjoint from conv phase)
  float* pre    = bufC;   // (B, 2560)
  float* synout = bufB;   // (B, 4096)
  float* hglu   = bufA;   // (B, 2048)
  float* preact = bufD;   // (B, 2048)
  float* ht1    = bufE;   // (2048, 256)
  float* ht2    = bufA;   // (2048, 64)

  // ---- stage weights to f16 (once per launch; deterministic) ----
  LAUNCH1D(f2h_kernel, 256 * 256, F(17), efc1h, 256 * 256);
  LAUNCH1D(f2h_kernel, RR * 128, F(19), efc2h, RR * 128);
  LAUNCH1D(f2h_kernel, RR * RR, F(21), kvh, RR * RR);
  LAUNCH1D(f2h_kernel, RR * RR, F(27) + (size_t)2 * RR * RR, wvh, RR * RR);
  LAUNCH1D(f2h_kernel, RR * RR, F(29), mouth, RR * RR);
  LAUNCH1D(f2h_kernel, 2 * NN * (RR + NN), F(31), synh, 2 * NN * (RR + NN));
  LAUNCH1D(f2h_kernel, 64 * NN, F(41), gA1h, 64 * NN);
  LAUNCH1D(f2h_kernel, 64 * NN, F(45), gO1h, 64 * NN);
  LAUNCH1D(f2h_kernel, NN * 256 * MM, F(35), nmw1h, NN * 256 * MM);
  for (int i = 0; i < 2; ++i) {
    int pb = 51 + i * 10;
    LAUNCH1D(f2h_kernel, 256 * SOO, F(pb + 0), tw1h[i], 256 * SOO);
    LAUNCH1D(f2h_kernel, 64 * 256, F(pb + 4), tw2h[i], 64 * 256);
  }

  // ---- encoder (once) ----
  LAUNCH1D(conv3x3_bn_relu, BB * 16 * 32 * 32, x, F(1), F(2), F(3), F(4), bufA, 3, 16, 32, 32, BB * 16 * 32 * 32);
  LAUNCH1D(conv3x3_bn_relu, BB * 32 * 32 * 32, bufA, F(5), F(6), F(7), F(8), bufB, 16, 32, 32, 32, BB * 32 * 32 * 32);
  LAUNCH1D(maxpool2_kernel, BB * 32 * 16 * 16, bufB, bufC, 32, 32, 32, BB * 32 * 16 * 16);
  LAUNCH1D(conv3x3_bn_relu, BB * 64 * 16 * 16, bufC, F(9), F(10), F(11), F(12), bufA, 32, 64, 16, 16, BB * 64 * 16 * 16);
  LAUNCH1D(maxpool2_kernel, BB * 64 * 8 * 8, bufA, bufD, 64, 16, 16, BB * 64 * 8 * 8);
  LAUNCH1D(conv3x3_bn_relu, BB * 64 * 8 * 8, bufD, F(13), F(14), F(15), F(16), bufE, 64, 64, 8, 8, BB * 64 * 8 * 8);
  LAUNCH1D(avgpool_flat_kernel, BB * 256, bufE, flat, BB * 256);
  gemm(stream, flat, 256, efc1h, 256, F(18), tmp256, 256, BB, 256, 256, 0);   // efc1
  LAUNCH1D(glu_kernel, BB * 128, tmp256, gluo, 128, BB * 128);
  gemm(stream, gluo, 128, efc2h, 128, F(20), feat, RR, BB, RR, 128, 0);        // efc2
  gemm(stream, feat, RR, kvh, RR, F(22), kvt, RR, BB, RR, RR, 0);              // kv linear
  layernorm_kernel<<<BB, 256, 0, stream>>>(kvt, kvb, F(23), F(24), RR, 0);     // kv LN
  // single-kv-token MHA: softmax over 1 token == 1 => attn = out_proj(Wv kv + bv), tick-invariant
  gemm(stream, kvb, RR, wvh, RR, F(28) + 2 * RR, vh, RR, BB, RR, RR, 0);
  gemm(stream, vh, RR, mouth, RR, F(30), attn, RR, BB, RR, RR, 0);
  LAUNCH1D(copy_to_pre_kernel, BB * RR, attn, pre, RR, 0, BB * RR);

  // ---- recurrent state init ----
  LAUNCH1D(bcast_rows_kernel, BB * NN, start_act, actb, NN, BB * NN);
  LAUNCH1D(trace_init_kernel, BB * NN * MM, start_trace, trace16, BB * NN * MM);
  hipMemsetAsync(aO, 0, (size_t)BB * SETS4 * SOO * sizeof(float), stream);
  hipMemsetAsync(bO, 0, (size_t)BB * SETS4 * SOO * sizeof(float), stream);

  // ---- tick loop ----
  for (int t = 0; t < TT; ++t) {
    // action-side gating (only gwA is live; sync_action -> q path is query-independent)
    gemm(stream, actb, NN, gA1h, NN, gA1_b, gh, 64, BB, 64, NN, 1);
    LAUNCH1D(gate_softmax_kernel, BB, gh, gA2_w, gA2_b, gwA_t + (size_t)t * BB * SETS4);
    // synapse
    LAUNCH1D(copy_to_pre_kernel, BB * NN, actb, pre, NN, RR, BB * NN);
    gemm(stream, pre, RR + NN, synh, RR + NN, syn_b, synout, 2 * NN, BB, 2 * NN, RR + NN, 0);
    LAUNCH1D(glu_kernel, BB * NN, synout, hglu, NN, BB * NN);
    layernorm_kernel<<<BB, 256, 0, stream>>>(hglu, preact, syn_ln_g, syn_ln_b, NN, 0);
    // linear trace: write newest at slot 32+t; neuron reads window [t+1, t+32]
    LAUNCH1D(trace_store_kernel, BB * NN, preact, trace16, 32 + t, BB * NN);
    neuron_kernel<<<NN * (BB / 16), 32, 0, stream>>>(trace16, nmw1h, nm_b1, nm_w2, nm_b2,
                                                     actb, t + 1);
    // out-side gating + synchronization
    gemm(stream, actb, NN, gO1h, NN, gO1_b, gh, 64, BB, 64, NN, 1);
    LAUNCH1D(gate_softmax_kernel, BB, gh, gO2_w, gO2_b, gwO_t + (size_t)t * BB * SETS4);
    LAUNCH1D(sync_out_kernel, BB * SOO, actb, idx_out_l, idx_out_r, decay_o,
             gwO_t + (size_t)t * BB * SETS4, aO, bO,
             syncO_t + (size_t)t * BB * SOO, SOO, BB * SOO);
  }

  // ---- task heads over all T*B rows ----
  const int rows = TT * BB;
  for (int i = 0; i < 2; ++i) {
    int pb = 51 + i * 10;
    int osz = (i == 0) ? 10 : 100;
    float* outp = (float*)d_out + ((i == 0) ? 0 : (size_t)TT * BB * 10);
    gemm(stream, syncO_t, SOO, tw1h[i], SOO, F(pb + 1), ht1, 256, rows, 256, SOO, 0);
    layernorm_kernel<<<rows, 256, 0, stream>>>(ht1, ht1, F(pb + 2), F(pb + 3), 256, 1);
    gemm(stream, ht1, 256, tw2h[i], 256, F(pb + 5), ht2, 64, rows, 64, 256, 0);
    layernorm_kernel<<<rows, 256, 0, stream>>>(ht2, ht2, F(pb + 6), F(pb + 7), 64, 1);
    LAUNCH1D(head_out_kernel, rows * osz, ht2, F(pb + 8), F(pb + 9), outp, osz, rows * osz);
  }

  balance_kernel<<<1, 32, 0, stream>>>(gwA_t, gwO_t, (float*)d_out + (size_t)TT * BB * 110);
}
